// PeriDynamicAttentionBlackwell_17643725652044
// MI455X (gfx1250) — matrix operand order repaired and checked by reading, not compiled, not run on hardware
//
#include <hip/hip_runtime.h>
#include <hip/hip_bf16.h>
#include <math.h>

// ---------------------------------------------------------------------------
// PeriDynamic attention for MI455X (gfx1250, wave32, WMMA).
//  Stage -1: W_pf = blockdiag_h(W_disp_h) @ W_fused composed  (tiny)
//  Stage 0: pos_feat = rel_pos_emb @ W_pos; zero-bias scratch (tiny)
//  Stage 1: P   = x @ W_pf            [2048x1024x512]         (bf16 WMMA GEMM)
//  Stage 2: val = x @ W_val + b_val   [2048x1024x1024]        (bf16 WMMA GEMM)
//  Stage 3: window stage: fused = P[t']-P[t]+b_fused, gelu, bond/damage,
//           softmax, weighted val sum (one wave per (b,h,t))
//  Stage 4: out = att @ W_cproj + b_cproj [2048x1024x1024]    (bf16 WMMA GEMM)
// Key algebra: strain@W_fused is linear in disp, so the per-tap 16x32 GEMM
// collapses into one big tensor-core GEMM; b_disp terms cancel in P[t']-P[t].
// ---------------------------------------------------------------------------

#define NH 16
#define HS 64
#define BD 16
#define HORIZON 64
#define CDIM 1024
#define PF2 32          // 2*BD fused width
#define PDIM (NH * PF2) // 512

typedef __attribute__((ext_vector_type(16))) __bf16 v16bf;
typedef __attribute__((ext_vector_type(8)))  float  v8f;

union FragAB {
    v16bf    v;
    unsigned u[8];
};

__device__ __forceinline__ unsigned f2bf_raw(float f) {
    unsigned u = __builtin_bit_cast(unsigned, f);
    return u + 0x7FFFu + ((u >> 16) & 1u);     // round-to-nearest-even (>>16)
}
__device__ __forceinline__ unsigned short f2bf(float f) {
    return (unsigned short)(f2bf_raw(f) >> 16);
}
__device__ __forceinline__ unsigned pack2bf(float lo, float hi) {
    return (f2bf_raw(lo) >> 16) | (f2bf_raw(hi) & 0xFFFF0000u);
}

__device__ __forceinline__ float gelu_exact(float z) {
    return 0.5f * z * (1.0f + erff(z * 0.70710678118654752f));
}

// ---------------------------------------------------------------------------
// Generic f32-in / f32-out GEMM with bf16 WMMA core.
// out[M,N] = A[M,K] @ W[K,N] + bias[N].  M % 64 == 0, N % 128 == 0, K % 32 == 0.
// Block: 128 threads = 4 waves. Block tile 64x128, wave tile 32x64 (2x4 WMMA).
// ---------------------------------------------------------------------------
#define BM 64
#define BN 128
#define BK 32
#define LPITCH 40   // ushort pitch (80B = 20 banks -> conflict-free row walk)

__global__ __launch_bounds__(128) void gemm_bf16_wmma(
    const float* __restrict__ A, const float* __restrict__ W,
    const float* __restrict__ bias, float* __restrict__ out,
    int M, int N, int K) {
    __shared__ unsigned short As[BM * LPITCH];   // A tile, row-major (m,k)
    __shared__ unsigned short Bs[BN * LPITCH];   // W tile, TRANSPOSED (n,k)

    const int tid  = threadIdx.x;
    const int lane = tid & 31;
    const int wave = tid >> 5;
    const int m0 = blockIdx.y * BM;
    const int n0 = blockIdx.x * BN;
    const int wm = (wave >> 1) * 32;   // wave row offset in block tile
    const int wn = (wave & 1) * 64;    // wave col offset in block tile
    const int sub = lane >> 4;         // half-wave (ISA lane-group)
    const int r   = lane & 15;

    v8f acc[2][4];
    #pragma unroll
    for (int mi = 0; mi < 2; ++mi)
        #pragma unroll
        for (int ni = 0; ni < 4; ++ni)
            acc[mi][ni] = v8f{0.f, 0.f, 0.f, 0.f, 0.f, 0.f, 0.f, 0.f};

    for (int kk = 0; kk < K; kk += BK) {
        // Stage A tile: 64x32 f32 as float4 (b128 loads), pack -> b64 LDS store.
        #pragma unroll
        for (int i = 0; i < (BM * BK) / (128 * 4); ++i) {   // 4 iters
            int idx  = i * 128 + tid;                        // over 512 float4
            int row  = idx >> 3;
            int col4 = (idx & 7) * 4;
            float4 v = *(const float4*)&A[(size_t)(m0 + row) * K + (kk + col4)];
            unsigned long long pk =
                (unsigned long long)pack2bf(v.x, v.y) |
                ((unsigned long long)pack2bf(v.z, v.w) << 32);
            *(unsigned long long*)&As[row * LPITCH + col4] = pk;
        }
        // Stage W tile transposed: float4 global loads, scattered b16 stores.
        #pragma unroll
        for (int i = 0; i < (BN * BK) / (128 * 4); ++i) {   // 8 iters
            int idx   = i * 128 + tid;                       // over 1024 float4
            int krow  = idx >> 5;
            int ncol4 = (idx & 31) * 4;
            float4 v = *(const float4*)&W[(size_t)(kk + krow) * N + (n0 + ncol4)];
            Bs[(ncol4 + 0) * LPITCH + krow] = f2bf(v.x);
            Bs[(ncol4 + 1) * LPITCH + krow] = f2bf(v.y);
            Bs[(ncol4 + 2) * LPITCH + krow] = f2bf(v.z);
            Bs[(ncol4 + 3) * LPITCH + krow] = f2bf(v.w);
        }
        __syncthreads();

        // Build fragments per ISA 7.12.2 lane layouts.
        FragAB afr[2], bfr[4];
        #pragma unroll
        for (int mi = 0; mi < 2; ++mi) {
            int row = wm + mi * 16 + r;
            #pragma unroll
            for (int p = 0; p < 8; ++p) {
                int k = ((p < 4) ? (2 * p) : (2 * p + 8)) + 8 * sub; // A: {0..7,16..23}/{8..15,24..31}
                afr[mi].u[p] = *(const unsigned*)&As[row * LPITCH + k];
            }
        }
        #pragma unroll
        for (int ni = 0; ni < 4; ++ni) {
            int col = wn + ni * 16 + r;
            #pragma unroll
            for (int p = 0; p < 8; ++p) {
                int k = ((p < 4) ? (2 * p) : (2 * p + 8)) + 8 * sub; // B: {0..7,16..23}/{8..15,24..31}
                bfr[ni].u[p] = *(const unsigned*)&Bs[col * LPITCH + k];
            }
        }
        #pragma unroll
        for (int mi = 0; mi < 2; ++mi)
            #pragma unroll
            for (int ni = 0; ni < 4; ++ni)
                acc[mi][ni] = __builtin_amdgcn_wmma_f32_16x16x32_bf16(
                    false, afr[mi].v, false, bfr[ni].v,
                    (short)0, acc[mi][ni], false, false);
        __syncthreads();
    }

    // D layout: VGPR j -> row j + 8*sub, col = lane&15.
    #pragma unroll
    for (int mi = 0; mi < 2; ++mi)
        #pragma unroll
        for (int ni = 0; ni < 4; ++ni) {
            int gc = n0 + wn + ni * 16 + r;
            float bv = bias[gc];
            #pragma unroll
            for (int j = 0; j < 8; ++j) {
                int gr = m0 + wm + mi * 16 + j + 8 * sub;
                out[(size_t)gr * N + gc] = acc[mi][ni][j] + bv;
            }
        }
}

// ---------------------------------------------------------------------------
// Stage -1: W_pf[c, h*32+j] = sum_kk W_disp[c, h*16+kk] * W_fused[kk, j]
// ---------------------------------------------------------------------------
__global__ void compose_wpf_kernel(const float* __restrict__ Wd,
                                   const float* __restrict__ Wf,
                                   float* __restrict__ Wpf) {
    int i = blockIdx.x * blockDim.x + threadIdx.x;   // over CDIM*PDIM
    if (i < CDIM * PDIM) {
        int c  = i >> 9;          // / PDIM
        int hj = i & (PDIM - 1);
        int h  = hj >> 5, j = hj & 31;
        float a = 0.f;
        #pragma unroll
        for (int kk = 0; kk < BD; ++kk)
            a += Wd[(size_t)c * (NH * BD) + h * BD + kk] * Wf[kk * PF2 + j];
        Wpf[i] = a;
    }
}

// ---------------------------------------------------------------------------
// Stage 0: pos_feat[64,16] = rel_pos_emb[64,16] @ W_pos[16,16]; zero-bias fill.
// ---------------------------------------------------------------------------
__global__ void posfeat_kernel(const float* __restrict__ rel,
                               const float* __restrict__ Wp,
                               float* __restrict__ pf,
                               float* __restrict__ zbias) {
    int i = blockIdx.x * blockDim.x + threadIdx.x;
    if (i < HORIZON * BD) {
        int d = i >> 4, j = i & 15;
        float a = 0.f;
        #pragma unroll
        for (int k = 0; k < BD; ++k) a += rel[d * BD + k] * Wp[k * BD + j];
        pf[i] = a;
    }
    if (i < PDIM) zbias[i] = 0.f;    // zero bias for the P GEMM
}

// ---------------------------------------------------------------------------
// Stage 3: fused window attention. One wave per (b,h,t); 2 taps per lane.
// P layout (b,t,h,32), val layout (b,t,h,HS), att layout (b,t,h,HS).
// fused = P[t'] - P[t] + b_fused (disp-bias contributions cancel; padded
// taps t'<0 are masked exactly like the reference cmask).
// ---------------------------------------------------------------------------
__global__ __launch_bounds__(128) void peri_window_kernel(
    const float* __restrict__ P, const float* __restrict__ val,
    const float* __restrict__ pf,
    const float* __restrict__ b_fused,
    const float* __restrict__ W_bond,  const float* __restrict__ b_bond,
    const float* __restrict__ W_dmg,   const float* __restrict__ b_dmg,
    float* __restrict__ att, int T) {
    const int lane = threadIdx.x & 31;
    const int wave = threadIdx.x >> 5;
    const int item = blockIdx.x * 4 + wave;      // over B*NH*T
    const int t  = item % T;
    const int bh = item / T;
    const int h  = bh % NH;
    const int b  = bh / NH;

    const float* pbase = P + (size_t)b * T * PDIM;
    float pc[PF2];
    {
        const float4* pc4 = (const float4*)(pbase + (size_t)t * PDIM + h * PF2);
        #pragma unroll
        for (int k = 0; k < PF2 / 4; ++k) {
            float4 v = pc4[k];
            pc[4 * k + 0] = v.x; pc[4 * k + 1] = v.y;
            pc[4 * k + 2] = v.z; pc[4 * k + 3] = v.w;
        }
    }

    float lg[2];
    #pragma unroll
    for (int q = 0; q < 2; ++q) {
        int d  = lane + 32 * q;
        int tp = t - (HORIZON - 1) + d;
        float logit = -INFINITY;
        if (tp >= 0) {
            const float4* pw4 =
                (const float4*)(pbase + (size_t)tp * PDIM + h * PF2);
            float pw[PF2];
            #pragma unroll
            for (int k = 0; k < PF2 / 4; ++k) {
                float4 v = pw4[k];
                pw[4 * k + 0] = v.x; pw[4 * k + 1] = v.y;
                pw[4 * k + 2] = v.z; pw[4 * k + 3] = v.w;
            }
            float bl = b_bond[0];
            float dl = b_dmg[0];
            #pragma unroll
            for (int j = 0; j < BD; ++j) {
                float fb = pw[j]      - pc[j]      + b_fused[j];
                float fd = pw[j + BD] - pc[j + BD] + b_fused[j + BD];
                bl += gelu_exact(fb + pf[d * BD + j]) * W_bond[j];
                dl += gelu_exact(fd) * W_dmg[j];
            }
            float dmg = 1.0f / (1.0f + expf(-dl));
            logit = bl - 10.0f * dmg;
        }
        lg[q] = logit;
    }

    // Softmax over 64 taps (2 per lane), wave32 reductions.
    float m = fmaxf(lg[0], lg[1]);
    #pragma unroll
    for (int off = 16; off; off >>= 1) m = fmaxf(m, __shfl_xor(m, off));
    float e0 = expf(lg[0] - m);
    float e1 = expf(lg[1] - m);
    float sum = e0 + e1;
    #pragma unroll
    for (int off = 16; off; off >>= 1) sum += __shfl_xor(sum, off);
    float inv = 1.0f / sum;
    float w0 = e0 * inv, w1 = e1 * inv;

    // out[hs] = sum_d w[d] * val[b, t-63+d, h, hs]; lanes cover hs (coalesced).
    const float* vbase = val + (size_t)b * T * CDIM;
    float a0 = 0.f, a1 = 0.f;
    for (int d = 0; d < HORIZON; ++d) {
        float wd = (d < 32) ? __shfl(w0, d) : __shfl(w1, d - 32);
        int tp = t - (HORIZON - 1) + d;
        if (tp >= 0) {
            const float* vr = vbase + (size_t)tp * CDIM + h * HS;
            a0 += wd * vr[lane];
            a1 += wd * vr[lane + 32];
        }
    }
    float* o = att + ((size_t)b * T + t) * CDIM + h * HS;
    o[lane]      = a0;
    o[lane + 32] = a1;
}

// ---------------------------------------------------------------------------
extern "C" void kernel_launch(void* const* d_in, const int* in_sizes, int n_in,
                              void* d_out, int out_size, void* d_ws, size_t ws_size,
                              hipStream_t stream) {
    const float* x       = (const float*)d_in[0];
    const float* W_disp  = (const float*)d_in[1];
    // d_in[2] = b_disp (cancels in P[t']-P[t]; padded taps are masked)
    const float* W_val   = (const float*)d_in[3];
    const float* b_val   = (const float*)d_in[4];
    const float* rel_pos = (const float*)d_in[5];
    const float* W_fused = (const float*)d_in[6];
    const float* b_fused = (const float*)d_in[7];
    const float* W_pos   = (const float*)d_in[8];
    const float* W_bond  = (const float*)d_in[9];
    const float* b_bond  = (const float*)d_in[10];
    const float* W_dmg   = (const float*)d_in[11];
    const float* b_dmg   = (const float*)d_in[12];
    const float* W_cproj = (const float*)d_in[13];
    const float* b_cproj = (const float*)d_in[14];
    float* out = (float*)d_out;

    const int T  = 1024;
    const int BT = in_sizes[0] / CDIM;   // B*T
    const int B  = BT / T;

    // Workspace layout (fp32)
    float* ws    = (float*)d_ws;
    float* P     = ws;                                 // BT * 512
    float* val   = P + (size_t)BT * PDIM;              // BT * 1024
    float* att   = val + (size_t)BT * CDIM;            // BT * 1024
    float* pf    = att + (size_t)BT * CDIM;            // 1024
    float* zbias = pf + HORIZON * BD;                  // 512
    float* Wpf   = zbias + PDIM;                       // 1024 * 512

    // Stage -1: compose W_pf = blockdiag(W_disp) @ W_fused
    compose_wpf_kernel<<<(CDIM * PDIM) / 256, 256, 0, stream>>>(
        W_disp, W_fused, Wpf);

    // Stage 0: pos_feat + zero bias scratch
    posfeat_kernel<<<4, 256, 0, stream>>>(rel_pos, W_pos, pf, zbias);

    // Stage 1: P = x @ W_pf               [BT x 1024] @ [1024 x 512]
    gemm_bf16_wmma<<<dim3(PDIM / BN, BT / BM), 128, 0, stream>>>(
        x, Wpf, zbias, P, BT, PDIM, CDIM);

    // Stage 2: val = x @ W_val + b_val    [BT x 1024] @ [1024 x 1024]
    gemm_bf16_wmma<<<dim3(CDIM / BN, BT / BM), 128, 0, stream>>>(
        x, W_val, b_val, val, BT, CDIM, CDIM);

    // Stage 3: fused window attention -> att (b,t,h,hs) == (B,T,C)
    peri_window_kernel<<<(B * NH * T) / 4, 128, 0, stream>>>(
        P, val, pf, b_fused, W_bond, b_bond, W_dmg, b_dmg, att, T);

    // Stage 4: out = att @ W_cproj + b_cproj
    gemm_bf16_wmma<<<dim3(CDIM / BN, BT / BM), 128, 0, stream>>>(
        att, W_cproj, b_cproj, out, BT, CDIM, CDIM);
}
